// SelfAttention_6451040878821
// MI455X (gfx1250) — matrix operand order repaired and checked by reading, not compiled, not run on hardware
//
#include <hip/hip_runtime.h>
#include <hip/hip_bf16.h>

#define B_ 4
#define C_ 128
#define N_ 4096   // 64*64
#define NCH_ (N_ / 32)

typedef __attribute__((ext_vector_type(16))) __bf16 v16bf;
typedef __attribute__((ext_vector_type(8)))  __bf16 v8bf;
typedef __attribute__((ext_vector_type(8)))  float  v8f;
typedef __attribute__((ext_vector_type(4)))  int    v4i;

#if defined(__HIP_DEVICE_COMPILE__) && __has_builtin(__builtin_amdgcn_global_load_async_to_lds_b128)
#define ASYNC_OK 1
#else
#define ASYNC_OK 0
#endif

#if ASYNC_OK
// Builtin signature (from probe diagnostic): (as1 int4* src, as3 int4* dst, Ii, Ii)
typedef __attribute__((address_space(1))) v4i as1_v4i;
typedef __attribute__((address_space(3))) v4i as3_v4i;
#endif

static __device__ __forceinline__ v16bf cat8(v8bf lo, v8bf hi) {
    return __builtin_shufflevector(lo, hi, 0,1,2,3,4,5,6,7,8,9,10,11,12,13,14,15);
}
static __device__ __forceinline__ v8f wmma_bf16(v16bf a, v16bf b, v8f c) {
    // D = A(16x32 bf16) * B(32x16 bf16) + C(16x16 f32)
    return __builtin_amdgcn_wmma_f32_16x16x32_bf16(false, a, false, b, (short)0, c,
                                                   false, false);
}

// 16-byte global -> LDS copy: async (ASYNCcnt) when available, else sync.
static __device__ __forceinline__ void stage16(const __bf16* g, __bf16* l) {
#if ASYNC_OK
    __builtin_amdgcn_global_load_async_to_lds_b128(
        (as1_v4i*)(size_t)g, (as3_v4i*)(unsigned int)(size_t)l, 0, 0);
#else
    *(uint4*)l = *(const uint4*)g;
#endif
}

// ---------------------------------------------------------------------------
// Kernel 1: x[B,C,N] f32 -> xt[B,N,C] bf16 (tiled LDS transpose, coalesced)
// ---------------------------------------------------------------------------
__global__ void k_transpose_bf16(const float* __restrict__ x, void* ws) {
    __bf16* xt = (__bf16*)ws;
    __shared__ float tile[32][33];
    int b  = blockIdx.z;
    int c0 = blockIdx.y * 32;
    int n0 = blockIdx.x * 32;
    const float* xb = x + (size_t)b * C_ * N_;
    for (int i = threadIdx.y; i < 32; i += 8)
        tile[i][threadIdx.x] = xb[(size_t)(c0 + i) * N_ + n0 + threadIdx.x];
    __syncthreads();
    __bf16* xtb = xt + (size_t)b * N_ * C_;
    for (int i = threadIdx.y; i < 32; i += 8)
        xtb[(size_t)(n0 + i) * C_ + c0 + threadIdx.x] = (__bf16)tile[threadIdx.x][i];
}

// ---------------------------------------------------------------------------
// Kernel 2: QKV projection via WMMA.
//   A-frag:  xt[n, c]  (M = pixel tile of 16, K = c in chunks of 32)
//   B-frag:  W^T[c_in, c_out] = W[c_out, c_in] (f32 -> bf16 on the fly)
//   Outputs: qt,kt as [B,N,C] bf16 ; v as [B,C,N] bf16
// ---------------------------------------------------------------------------
__global__ void k_qkv_project(void* ws,
                              const float* __restrict__ wq, const float* __restrict__ bq,
                              const float* __restrict__ wk, const float* __restrict__ bk,
                              const float* __restrict__ wv, const float* __restrict__ bv) {
    __bf16* xt = (__bf16*)ws;
    __bf16* qt = xt + (size_t)B_ * N_ * C_;
    __bf16* kt = qt + (size_t)B_ * N_ * C_;
    __bf16* vb = kt + (size_t)B_ * N_ * C_;

    const int lane = threadIdx.x & 31;
    const int wave = threadIdx.x >> 5;
    const int wtile = blockIdx.x * 4 + wave;          // 0..1023
    const int b     = wtile >> 8;                     // wtile / 256
    const int nbase = (wtile & 255) << 4;             // 16-pixel tile
    const int col   = lane & 15;
    const int half  = lane >> 4;
    const int h8    = half * 8;

    // A fragments: 16 pixels x K=128, split into 4 chunks of K=32
    const __bf16* arow = xt + ((size_t)b * N_ + nbase + col) * C_;
    v16bf af[4];
    #pragma unroll
    for (int kk = 0; kk < 4; ++kk) {
        v8bf lo = *(const v8bf*)(arow + kk * 32 + half * 16);
        v8bf hi = *(const v8bf*)(arow + kk * 32 + half * 16 + 8);
        af[kk] = cat8(lo, hi);
    }

    const float* Ws[3] = { wq, wk, wv };
    const float* Bs[3] = { bq, bk, bv };

    for (int w = 0; w < 3; ++w) {
        const float* W = Ws[w];
        #pragma unroll
        for (int t = 0; t < 8; ++t) {               // 8 tiles of 16 output channels
            const int cbase = t * 16;
            v8f acc = {};
            #pragma unroll
            for (int kk = 0; kk < 4; ++kk) {
                const float* wrow = W + (size_t)(cbase + col) * C_ + kk * 32 + half * 16;
                const float4* w4 = (const float4*)wrow;
                float4 f0 = w4[0], f1 = w4[1], f2 = w4[2], f3 = w4[3];
                v16bf wbf;
                wbf[0]=(__bf16)f0.x; wbf[1]=(__bf16)f0.y; wbf[2]=(__bf16)f0.z; wbf[3]=(__bf16)f0.w;
                wbf[4]=(__bf16)f1.x; wbf[5]=(__bf16)f1.y; wbf[6]=(__bf16)f1.z; wbf[7]=(__bf16)f1.w;
                wbf[8]=(__bf16)f2.x; wbf[9]=(__bf16)f2.y; wbf[10]=(__bf16)f2.z; wbf[11]=(__bf16)f2.w;
                wbf[12]=(__bf16)f3.x; wbf[13]=(__bf16)f3.y; wbf[14]=(__bf16)f3.z; wbf[15]=(__bf16)f3.w;
                acc = wmma_bf16(af[kk], wbf, acc);
            }
            const float bias = Bs[w][cbase + col];
            #pragma unroll
            for (int r = 0; r < 8; ++r) acc[r] += bias;

            if (w < 2) {
                // q,k -> [B,N,C] bf16 (D rows = pixels, D cols = channels)
                __bf16* dst = (w == 0 ? qt : kt) + (size_t)b * N_ * C_;
                #pragma unroll
                for (int r = 0; r < 8; ++r)
                    dst[(size_t)(nbase + h8 + r) * C_ + cbase + col] = (__bf16)acc[r];
            } else {
                // v -> [B,C,N] bf16: 8 consecutive pixels per lane -> one 16B store
                v8bf pk;
                #pragma unroll
                for (int r = 0; r < 8; ++r) pk[r] = (__bf16)acc[r];
                *(v8bf*)(vb + (size_t)b * C_ * N_ + (size_t)(cbase + col) * N_ + nbase + h8) = pk;
            }
        }
    }
}

// ---------------------------------------------------------------------------
// Kernel 3: flash attention with double-buffered (async) K/V staging in LDS.
// Block = 4 waves; each wave owns 16 query rows. Per 32-col chunk:
//   - block stages K (32x128 bf16) + V (128x32 bf16) into LDS (8 async b128
//     per wave, ASYNCcnt-tracked, overlapped with compute of previous chunk)
//   - 8 WMMA for S (K=C=128), f32 online softmax (shfl_xor row reductions),
//     P restaged via LDS (D-layout -> A-layout), 8 WMMA for P*V.
// ---------------------------------------------------------------------------
__global__ void k_attention(void* ws, float* __restrict__ out) {
    __bf16* xt = (__bf16*)ws;
    __bf16* qt = xt + (size_t)B_ * N_ * C_;
    __bf16* kt = qt + (size_t)B_ * N_ * C_;
    __bf16* vb = kt + (size_t)B_ * N_ * C_;

    // Padded rows keep 32B alignment for v16bf reads and stagger LDS banks.
    __shared__ __bf16 Kbuf[2][32][144];   // [m][c], row stride 288B
    __shared__ __bf16 Vbuf[2][128][48];   // [c][m], row stride  96B
    __shared__ __bf16 Pbuf[4][16][32];    // per-wave P staging

    const int tid  = threadIdx.x;
    const int lane = tid & 31;
    const int wave = tid >> 5;
    const int wtile = blockIdx.x * 4 + wave;          // 0..1023
    const int b     = wtile >> 8;
    const int nbase = (wtile & 255) << 4;
    const int col   = lane & 15;
    const int half  = lane >> 4;
    const int h8    = half * 8;

    const __bf16* ktb = kt + (size_t)b * N_ * C_;
    const __bf16* vbb = vb + (size_t)b * C_ * N_;

    // Q A-fragments (held for whole kernel): 16 rows x K=128
    const __bf16* qrow = qt + ((size_t)b * N_ + nbase + col) * C_;
    v16bf qf[4];
    #pragma unroll
    for (int kk = 0; kk < 4; ++kk) {
        v8bf lo = *(const v8bf*)(qrow + kk * 32 + half * 16);
        v8bf hi = *(const v8bf*)(qrow + kk * 32 + half * 16 + 8);
        qf[kk] = cat8(lo, hi);
    }

    v8f zero = {};
    v8f o[8];                                         // 16 rows x 128 channels, f32
    #pragma unroll
    for (int t = 0; t < 8; ++t) o[t] = zero;
    float m_i[8], l_i[8];
    #pragma unroll
    for (int r = 0; r < 8; ++r) { m_i[r] = -3.0e38f; l_i[r] = 0.0f; }

    // --- block-wide staging of one 32-column K/V chunk into LDS buffer `buf`
    auto stage_chunk = [&](int mbase, int buf) {
        #pragma unroll
        for (int j = 0; j < 4; ++j) {                 // K: 512 x 16B segments
            int s = tid + 128 * j;
            int m = s >> 4, cs = (s & 15) * 8;
            stage16(ktb + (size_t)(mbase + m) * C_ + cs, &Kbuf[buf][m][cs]);
        }
        #pragma unroll
        for (int j = 0; j < 4; ++j) {                 // V: 512 x 16B segments
            int s = tid + 128 * j;
            int c = s >> 2, ms = (s & 3) * 8;
            stage16(vbb + (size_t)c * N_ + mbase + ms, &Vbuf[buf][c][ms]);
        }
    };

    stage_chunk(0, 0);                                // prologue: chunk 0

    for (int mc = 0; mc < NCH_; ++mc) {
        const int cur = mc & 1;
        if (mc + 1 < NCH_) {
            stage_chunk((mc + 1) * 32, cur ^ 1);      // overlap next chunk copy
#if ASYNC_OK
            asm volatile("s_wait_asynccnt 0x8" ::: "memory");  // cur chunk landed
#endif
        } else {
#if ASYNC_OK
            asm volatile("s_wait_asynccnt 0x0" ::: "memory");
#endif
        }
        __syncthreads();                              // all waves' deposits visible

        // --- S = Q^T K : two 16x16 f32 tiles from LDS K fragments
        v8f s0 = zero, s1 = zero;
        const __bf16* kb0 = &Kbuf[cur][col][half * 16];
        const __bf16* kb1 = &Kbuf[cur][16 + col][half * 16];
        #pragma unroll
        for (int kk = 0; kk < 4; ++kk)
            s0 = wmma_bf16(qf[kk], *(const v16bf*)(kb0 + kk * 32), s0);
        #pragma unroll
        for (int kk = 0; kk < 4; ++kk)
            s1 = wmma_bf16(qf[kk], *(const v16bf*)(kb1 + kk * 32), s1);

        // --- online softmax (rows r + half*8; reduce across 16-lane halves)
        #pragma unroll
        for (int r = 0; r < 8; ++r) {
            float a = s0[r], c = s1[r];
            float cm = fmaxf(a, c);
            #pragma unroll
            for (int off = 8; off >= 1; off >>= 1)
                cm = fmaxf(cm, __shfl_xor(cm, off, 32));
            float mnew  = fmaxf(m_i[r], cm);
            float scale = __expf(m_i[r] - mnew);
            m_i[r] = mnew;
            float p0 = __expf(a - mnew);
            float p1 = __expf(c - mnew);
            float rs = p0 + p1;
            #pragma unroll
            for (int off = 8; off >= 1; off >>= 1)
                rs += __shfl_xor(rs, off, 32);
            l_i[r] = l_i[r] * scale + rs;
            #pragma unroll
            for (int t = 0; t < 8; ++t) o[t][r] *= scale;
            Pbuf[wave][h8 + r][col]      = (__bf16)p0;
            Pbuf[wave][h8 + r][col + 16] = (__bf16)p1;
        }

        // --- rebuild P as A-fragment (LDS ops are in-order within a wave)
        v8bf plo = *(const v8bf*)&Pbuf[wave][col][half * 16];
        v8bf phi = *(const v8bf*)&Pbuf[wave][col][half * 16 + 8];
        v16bf pf = cat8(plo, phi);

        // --- O += P (16x32) * V^T (32x16 per channel tile) from LDS V frags
        #pragma unroll
        for (int t = 0; t < 8; ++t) {
            v16bf vf = *(const v16bf*)&Vbuf[cur][t * 16 + col][half * 16];
            o[t] = wmma_bf16(pf, vf, o[t]);
        }
        __syncthreads();   // protect `cur` (refilled at top of next iteration)
    }

    // --- epilogue: normalize and store out[b, c, n] (8 consecutive n per lane)
    float inv[8];
    #pragma unroll
    for (int r = 0; r < 8; ++r) inv[r] = 1.0f / l_i[r];
    float* ob = out + (size_t)b * C_ * N_;
    #pragma unroll
    for (int t = 0; t < 8; ++t) {
        float* dst = ob + (size_t)(t * 16 + col) * N_ + nbase + h8;
        float4 f0 = make_float4(o[t][0] * inv[0], o[t][1] * inv[1],
                                o[t][2] * inv[2], o[t][3] * inv[3]);
        float4 f1 = make_float4(o[t][4] * inv[4], o[t][5] * inv[5],
                                o[t][6] * inv[6], o[t][7] * inv[7]);
        *(float4*)dst       = f0;
        *(float4*)(dst + 4) = f1;
    }
}

// ---------------------------------------------------------------------------
extern "C" void kernel_launch(void* const* d_in, const int* in_sizes, int n_in,
                              void* d_out, int out_size, void* d_ws, size_t ws_size,
                              hipStream_t stream) {
    (void)in_sizes; (void)n_in; (void)out_size; (void)ws_size;
    const float* x  = (const float*)d_in[0];
    const float* wq = (const float*)d_in[1];
    const float* bq = (const float*)d_in[2];
    const float* wk = (const float*)d_in[3];
    const float* bk = (const float*)d_in[4];
    const float* wv = (const float*)d_in[5];
    const float* bv = (const float*)d_in[6];
    float* out = (float*)d_out;

    // ws layout (bf16): xt[B,N,C] | qt[B,N,C] | kt[B,N,C] | v[B,C,N]  = 16 MB
    dim3 gT(N_ / 32, C_ / 32, B_), bT(32, 8, 1);
    k_transpose_bf16<<<gT, bT, 0, stream>>>(x, d_ws);

    // 1024 wave-tiles of 16 pixels, 4 waves (128 threads) per block
    k_qkv_project<<<(B_ * N_ / 16) / 4, 128, 0, stream>>>(d_ws, wq, bq, wk, bk, wv, bv);

    // 1024 wave-tiles of 16 query rows, 4 waves per block
    k_attention<<<(B_ * N_ / 16) / 4, 128, 0, stream>>>(d_ws, out);
}
